// MSAColumnGlobalAttention_46918222742019
// MI455X (gfx1250) — matrix operand order repaired and checked by reading, not compiled, not run on hardware
//
#include <hip/hip_runtime.h>
#include <stdint.h>

// MSAColumnGlobalAttention, MI455X (gfx1250, wave32, WMMA).
// One workgroup (1024 thr = 32 waves) per column r; entire column's
// LayerNormed X lives in LDS as f16 (256KB). Single read of m, single
// write of out => memory-roofline design (~402MB @ 23.3TB/s).

typedef __attribute__((ext_vector_type(16))) _Float16 v16h;
typedef __attribute__((ext_vector_type(8)))  float    v8f;

union V16H { v16h v; uint32_t u[8]; _Float16 h[16]; };
union V8F  { v8f  v; float f[8]; };
union H2   { _Float16 h[2]; uint32_t u; };

#define S_DIM 2048
#define R_DIM 384
#define C_DIM 64
#define HC_DIM 64
#define NWAVE 32
// LDS carve (bytes)
#define XH_BYTES   (S_DIM * C_DIM * 2)            // 262144
#define PH_BYTES   (S_DIM * 8 * 2)                // 32768
#define SMEM_BYTES (XH_BYTES + PH_BYTES + 8192 + 8192 + 2048 + (512+64+64+512+512+64)*4)

__device__ __forceinline__ float wave_sum32(float v) {
#pragma unroll
  for (int d = 16; d > 0; d >>= 1) v += __shfl_xor(v, d, 32);
  return v;
}

// A-fragment of a row-major [*,64] f16 LDS matrix (16x32 f16 A, contiguous k = 16*(lane/16) + s)
__device__ __forceinline__ V16H load_a_rows(const uint32_t* X32, int rowBase, int kb, int lane) {
  V16H a;
  const int mrow = rowBase + (lane & 15);
  const int h4   = (lane >> 4) << 3;
  const uint32_t* p = X32 + mrow * 32 + (kb >> 1) + h4;
  a.u[0] = p[0]; a.u[1] = p[1]; a.u[2] = p[2]; a.u[3] = p[3];
  a.u[4] = p[4]; a.u[5] = p[5]; a.u[6] = p[6]; a.u[7] = p[7];
  return a;
}

// B-fragment from a fragment-ordered LDS weight buffer: [(kc*NT+nt)*8+i]*32+lane
__device__ __forceinline__ V16H load_b_frag(const uint32_t* buf, int kc, int nt, int NT, int lane) {
  V16H b;
  const uint32_t* p = buf + ((kc * NT + nt) * 8) * 32 + lane;
#pragma unroll
  for (int i = 0; i < 8; ++i) b.u[i] = p[i * 32];
  return b;
}

// A-fragment of X^T (M=c, K=s): gathers column c of Xh (16-bit A layout)
__device__ __forceinline__ V16H load_a_xt(const _Float16* Xh, int ctBase, int kb, int lane) {
  V16H a;
  const int c = ctBase + (lane & 15);
  const int hf = lane >> 4;
#pragma unroll
  for (int i = 0; i < 8; ++i) {
    const int k0 = kb + hf * 16 + 2 * i;
    a.h[2 * i]     = Xh[k0 * 64 + c];
    a.h[2 * i + 1] = Xh[(k0 + 1) * 64 + c];
  }
  return a;
}

// B-fragment of P (K=s, N=h; 32x16 f16 B layout; cols 8..15 zero)
__device__ __forceinline__ V16H load_b_p(const _Float16* Ph, int kb, int lane) {
  V16H b;
  const int n = lane & 15;
  const int hf = lane >> 4;
  if (n < 8) {
#pragma unroll
    for (int i = 0; i < 8; ++i) {
      const int k0 = kb + hf * 16 + 2 * i;
      b.h[2 * i]     = Ph[k0 * 8 + n];
      b.h[2 * i + 1] = Ph[(k0 + 1) * 8 + n];
    }
  } else {
#pragma unroll
    for (int i = 0; i < 8; ++i) b.u[i] = 0u;
  }
  return b;
}

__global__ __launch_bounds__(1024) void msa_col_global_attn_kernel(
    const float* __restrict__ m, const float* __restrict__ ln_w, const float* __restrict__ ln_b,
    const float* __restrict__ Wq, const float* __restrict__ Wk, const float* __restrict__ Wv,
    const float* __restrict__ Wg, const float* __restrict__ bg,
    const float* __restrict__ Wo, const float* __restrict__ bo, float* __restrict__ out) {
  extern __shared__ char smem[];
  _Float16* Xh  = (_Float16*)smem;                        // [2048][64] LN'd X (later overwritten by Y)
  _Float16* Ph  = (_Float16*)(smem + XH_BYTES);           // [2048][8] exp(logit - max)
  uint32_t* WgB = (uint32_t*)(smem + XH_BYTES + PH_BYTES);// 2048 dw, B-frag order
  uint32_t* WoB = WgB + 2048;                             // 2048 dw
  uint32_t* WlB = WoB + 2048;                             // 512 dw (N-tile 0 only)
  float* pxT  = (float*)(WlB + 512);                      // [64][8] = P^T X -> px^T
  float* qsum = pxT + 512;                                // [64]
  float* qv   = qsum + 64;                                // [64] q flat
  float* scrM = qv + 64;                                  // [32][16] per-wave max
  float* scrS = scrM + 512;                               // [32][16] per-wave sumexp
  float* obc  = scrS + 512;                               // [64] o[h,ch]/denom broadcast
  uint32_t* X32 = (uint32_t*)Xh;

  const int tid  = threadIdx.x;
  const int lane = tid & 31;
  const int w    = tid >> 5;
  const int n    = lane & 15;
  const int hf   = lane >> 4;
  const int r    = blockIdx.x;

  const v8f zero8 = {};

  // ---- Init: zero accumulators, stage Wg/Wo into B-fragment order (f16) ----
  if (tid < 64)  qsum[tid] = 0.0f;
  if (tid < 512) pxT[tid]  = 0.0f;
#pragma unroll
  for (int e = 0; e < 2; ++e) {
    const int idx = tid * 2 + e;               // 2048 fragment dwords each
    const int l2 = idx & 31, fi = (idx >> 5) & 7, nt = (idx >> 8) & 3, kc = idx >> 10;
    const int nn = l2 & 15, hh = l2 >> 4;
    const int k0 = kc * 32 + hh * 16 + 2 * fi;
    const int uc = nt * 16 + nn;
    H2 pg; pg.h[0] = (_Float16)Wg[k0 * 64 + uc]; pg.h[1] = (_Float16)Wg[(k0 + 1) * 64 + uc];
    WgB[idx] = pg.u;
    H2 po; po.h[0] = (_Float16)Wo[k0 * 64 + uc]; po.h[1] = (_Float16)Wo[(k0 + 1) * 64 + uc];
    WoB[idx] = po.u;
  }
  __syncthreads();

  // ---- Phase 1: LayerNorm (f32 stats via shuffles), store f16 X, pooled-q sums ----
  const float lw0 = ln_w[2 * lane], lw1 = ln_w[2 * lane + 1];
  const float lb0 = ln_b[2 * lane], lb1 = ln_b[2 * lane + 1];
  float qa0 = 0.0f, qa1 = 0.0f;
  for (int t = 0; t < 64; ++t) {
    const int s = w * 64 + t;
    const float2 xv = *(const float2*)(m + (((size_t)s * R_DIM + r) << 6) + 2 * lane);
    const float mu = wave_sum32(xv.x + xv.y) * (1.0f / 64.0f);
    const float d0 = xv.x - mu, d1 = xv.y - mu;
    const float var = wave_sum32(d0 * d0 + d1 * d1) * (1.0f / 64.0f);
    const float rs = rsqrtf(var + 1e-5f);
    const float x0 = d0 * rs * lw0 + lb0;
    const float x1 = d1 * rs * lw1 + lb1;
    qa0 += x0; qa1 += x1;
    H2 pk; pk.h[0] = (_Float16)x0; pk.h[1] = (_Float16)x1;
    X32[s * 32 + lane] = pk.u;
  }
  atomicAdd(&qsum[2 * lane], qa0);
  atomicAdd(&qsum[2 * lane + 1], qa1);
  __syncthreads();

  // ---- Phase 2 (wave 0): q = (sum/(S+eps)) Wq * 8^-0.5 ; Wl = Wk q^T in B-layout ----
  if (w == 0) {
    const float inv_pool = 1.0f / (2048.0f + 1e-10f);
    const float rs8 = 0.35355339059327373f;  // 8^-0.5
#pragma unroll
    for (int rep = 0; rep < 2; ++rep) {
      const int u = lane + rep * 32;
      float acc = 0.0f;
      for (int c = 0; c < 64; ++c) acc += qsum[c] * Wq[c * 64 + u];
      qv[u] = acc * inv_pool * rs8;
    }
    asm volatile("s_wait_dscnt 0" ::: "memory");
#pragma unroll
    for (int e = 0; e < 16; ++e) {           // 512 WlB dwords, 16 per lane
      const int fi = e & 7, kc = e >> 3;
      const int k0 = kc * 32 + hf * 16 + 2 * fi;
      H2 pk; pk.u = 0u;
      if (n < 8) {
        float lo = 0.0f, hi = 0.0f;
#pragma unroll
        for (int j = 0; j < 8; ++j) {
          const float qj = qv[n * 8 + j];
          lo += Wk[k0 * 8 + j] * qj;
          hi += Wk[(k0 + 1) * 8 + j] * qj;
        }
        pk.h[0] = (_Float16)lo; pk.h[1] = (_Float16)hi;
      }
      WlB[(kc * 8 + fi) * 32 + lane] = pk.u;
    }
  }
  __syncthreads();

  // ---- Phase 3: logits = X * Wl (WMMA), softmax over s, P = exp(l - max) ----
  V8F cfr[4];
  {
    const V16H bl0 = load_b_frag(WlB, 0, 0, 1, lane);
    const V16H bl1 = load_b_frag(WlB, 1, 0, 1, lane);
#pragma unroll
    for (int t = 0; t < 4; ++t) {
      const int rb = (w * 4 + t) * 16;
      const V16H a0 = load_a_rows(X32, rb, 0, lane);
      const V16H a1 = load_a_rows(X32, rb, 32, lane);
      v8f c = zero8;
      c = __builtin_amdgcn_wmma_f32_16x16x32_f16(false, a0.v, false, bl0.v, (short)0, c, false, false);
      c = __builtin_amdgcn_wmma_f32_16x16x32_f16(false, a1.v, false, bl1.v, (short)0, c, false, false);
      cfr[t].v = c;
    }
  }
  float lm = -3.0e38f;
#pragma unroll
  for (int t = 0; t < 4; ++t)
#pragma unroll
    for (int i = 0; i < 8; ++i) lm = fmaxf(lm, cfr[t].f[i]);
  lm = fmaxf(lm, __shfl_xor(lm, 16, 32));
  if (lane < 16) scrM[w * 16 + lane] = lm;
  __syncthreads();
  float gm = -3.0e38f;
  for (int ww = 0; ww < NWAVE; ++ww) gm = fmaxf(gm, scrM[ww * 16 + n]);
  float ls = 0.0f;
#pragma unroll
  for (int t = 0; t < 4; ++t)
#pragma unroll
    for (int i = 0; i < 8; ++i) {
      const float e = __expf(cfr[t].f[i] - gm);
      cfr[t].f[i] = e;
      ls += e;
    }
  if (n < 8) {
#pragma unroll
    for (int t = 0; t < 4; ++t) {
      const int rb = (w * 4 + t) * 16 + hf * 8;
#pragma unroll
      for (int i = 0; i < 8; ++i) Ph[(rb + i) * 8 + n] = (_Float16)cfr[t].f[i];
    }
  }
  ls += __shfl_xor(ls, 16, 32);
  if (lane < 16) scrS[w * 16 + lane] = ls;
  __syncthreads();

  // ---- Phase 4: px^T = X^T * P (WMMA over K=s), LDS-atomic combine over K-groups ----
  {
    const int ct = (w & 3) * 16;   // c-tile
    const int kg = w >> 2;         // 8 K-groups x 8 chunks of 32
    V8F cacc; cacc.v = zero8;
#pragma unroll
    for (int cc = 0; cc < 8; ++cc) {
      const int kb = (kg * 8 + cc) * 32;
      const V16H a = load_a_xt(Xh, ct, kb, lane);
      const V16H b = load_b_p(Ph, kb, lane);
      cacc.v = __builtin_amdgcn_wmma_f32_16x16x32_f16(false, a.v, false, b.v, (short)0, cacc.v, false, false);
    }
    if (n < 8) {
#pragma unroll
      for (int i = 0; i < 8; ++i) {
        const int cr = ct + hf * 8 + i;
        atomicAdd(&pxT[cr * 8 + n], cacc.f[i]);
      }
    }
  }
  __syncthreads();

  // ---- Phase 5 (wave 0): o[h,j] = (px^T^T Wv)[h,j] / sumexp[h]  -> obc[u] ----
  if (w == 0) {
#pragma unroll
    for (int rep = 0; rep < 2; ++rep) {
      const int u = lane + rep * 32;
      const int h = u >> 3, j = u & 7;
      float acc = 0.0f;
      for (int c = 0; c < 64; ++c) acc += pxT[c * 8 + h] * Wv[c * 8 + j];
      float gs = 0.0f;
      for (int ww = 0; ww < NWAVE; ++ww) gs += scrS[ww * 16 + h];
      obc[u] = acc / gs;
    }
  }
  __syncthreads();

  // ---- Phase 6: per M-tile: G = sigmoid(X Wg + bg); Y = G*obc (in place over X);
  //               out = Y Wo + bo  -> global (transposed [S,R,C]) ----
#pragma unroll 1
  for (int t = 0; t < 4; ++t) {
    const int rb = (w * 4 + t) * 16;
    const V16H ax0 = load_a_rows(X32, rb, 0, lane);
    const V16H ax1 = load_a_rows(X32, rb, 32, lane);
#pragma unroll
    for (int nt = 0; nt < 4; ++nt) {
      const V16H b0 = load_b_frag(WgB, 0, nt, 4, lane);
      const V16H b1 = load_b_frag(WgB, 1, nt, 4, lane);
      V8F c; c.v = zero8;
      c.v = __builtin_amdgcn_wmma_f32_16x16x32_f16(false, ax0.v, false, b0.v, (short)0, c.v, false, false);
      c.v = __builtin_amdgcn_wmma_f32_16x16x32_f16(false, ax1.v, false, b1.v, (short)0, c.v, false, false);
      const int uc = nt * 16 + n;
      const float bgv = bg[uc];
      const float ov  = obc[uc];
#pragma unroll
      for (int i = 0; i < 8; ++i) {
        const float g = 1.0f / (1.0f + __expf(-(c.f[i] + bgv)));
        Xh[(rb + hf * 8 + i) * 64 + uc] = (_Float16)(g * ov);
      }
    }
    asm volatile("s_wait_dscnt 0" ::: "memory");   // in-place X -> Y turnaround
    const V16H ay0 = load_a_rows(X32, rb, 0, lane);
    const V16H ay1 = load_a_rows(X32, rb, 32, lane);
#pragma unroll
    for (int nt = 0; nt < 4; ++nt) {
      const V16H b0 = load_b_frag(WoB, 0, nt, 4, lane);
      const V16H b1 = load_b_frag(WoB, 1, nt, 4, lane);
      V8F c; c.v = zero8;
      c.v = __builtin_amdgcn_wmma_f32_16x16x32_f16(false, ay0.v, false, b0.v, (short)0, c.v, false, false);
      c.v = __builtin_amdgcn_wmma_f32_16x16x32_f16(false, ay1.v, false, b1.v, (short)0, c.v, false, false);
      const int cc = nt * 16 + n;
      const float bov = bo[cc];
#pragma unroll
      for (int i = 0; i < 8; ++i) {
        const int s = rb + hf * 8 + i;
        out[(((size_t)s * R_DIM + r) << 6) + cc] = c.f[i] + bov;
      }
    }
  }
}

extern "C" void kernel_launch(void* const* d_in, const int* in_sizes, int n_in,
                              void* d_out, int out_size, void* d_ws, size_t ws_size,
                              hipStream_t stream) {
  (void)in_sizes; (void)n_in; (void)d_ws; (void)ws_size; (void)out_size;
  const float* m    = (const float*)d_in[0];
  const float* ln_w = (const float*)d_in[1];
  const float* ln_b = (const float*)d_in[2];
  const float* Wq   = (const float*)d_in[3];
  const float* Wk   = (const float*)d_in[4];
  const float* Wv   = (const float*)d_in[5];
  const float* Wg   = (const float*)d_in[6];
  const float* bg   = (const float*)d_in[7];
  const float* Wo   = (const float*)d_in[8];
  const float* bo   = (const float*)d_in[9];
  float* outp = (float*)d_out;

  (void)hipFuncSetAttribute((const void*)msa_col_global_attn_kernel,
                            hipFuncAttributeMaxDynamicSharedMemorySize, SMEM_BYTES);
  msa_col_global_attn_kernel<<<dim3(R_DIM), dim3(1024), SMEM_BYTES, stream>>>(
      m, ln_w, ln_b, Wq, Wk, Wv, Wg, bg, Wo, bo, outp);
}
